// GPT_76029511073868
// MI455X (gfx1250) — compile-verified, matrix operand order checked by reading, not run on hardware
//
#include <hip/hip_runtime.h>
#include <hip/hip_bf16.h>

// ---------------------------------------------------------------------------
// GPT forward for gfx1250 (CDNA5): bf16 WMMA GEMMs + flash attention.
// B=2, T=1024, D=768, H=12, HS=64, L=6, V=50257
// ---------------------------------------------------------------------------

#define GB  2
#define GT  1024
#define GD  768
#define GH  12
#define GHS 64
#define GL  6
#define GV  50257
#define GM  (GB * GT)          // 2048 rows of the token stream
#define GFF (4 * GD)           // 3072

#define BM 128
#define BN 128
#define BK 32

typedef __attribute__((ext_vector_type(16))) __bf16 v16bf;
typedef __attribute__((ext_vector_type(8)))  float  v8f;

// ---------------------------------------------------------------------------
// Embedding: x[b,t,:] = tok_emb[idx[b,t],:] + pos_emb[t,:]
// ---------------------------------------------------------------------------
__global__ __launch_bounds__(256) void embed_kernel(
    const int* __restrict__ idx, const float* __restrict__ tok,
    const float* __restrict__ pos, float* __restrict__ x)
{
    int i = blockIdx.x * 256 + threadIdx.x;
    if (i >= GM * GD) return;
    int d  = i % GD;
    int bt = i / GD;
    int t  = bt % GT;
    int tk = idx[bt];
    x[i] = tok[(size_t)tk * GD + d] + pos[(size_t)t * GD + d];
}

// ---------------------------------------------------------------------------
// GEMM: C[M,N] = act(A[M,K] @ B[K,N] + bias)
//   fp32 A/B converted to bf16 LDS tiles; v_wmma_f32_16x16x32_bf16.
//   Block: 256 thr = 8 waves arranged 4(M) x 2(N); block tile 128x128, BK=32.
//   Per wave: 32x64 sub-tile = 2(M) x 4(N) WMMA tiles = 8 wmma per K-step.
//   headMajor: B element (k, j) at  (j>>6)*(K*64) + k*64 + (j&63)   [wq/wk/wv]
// ---------------------------------------------------------------------------
__global__ __launch_bounds__(256) void gemm_kernel(
    const float* __restrict__ A, const float* __restrict__ B,
    const float* __restrict__ bias, float* __restrict__ C,
    int M, int N, int K, int headMajor, int relu)
{
    __shared__ __bf16 As[BM][BK];        // [m][k]  8 KB
    __shared__ __bf16 Bs[BN][BK];        // [n][k]  8 KB (K-transposed)

    const int tid  = threadIdx.x;
    const int lane = tid & 31;
    const int wid  = tid >> 5;
    const int wm   = wid >> 1;           // 0..3  (32-row strip)
    const int wn   = wid & 1;            // 0..1  (64-col strip)
    const int m0   = blockIdx.y * BM;
    const int n0   = blockIdx.x * BN;
    const int half = lane >> 4;          // K-half select
    const int r    = lane & 15;
    const int kb   = half * 8;

    v8f acc[2][4];
#pragma unroll
    for (int mi = 0; mi < 2; ++mi)
#pragma unroll
        for (int ni = 0; ni < 4; ++ni) { v8f z = {}; acc[mi][ni] = z; }

    for (int k0 = 0; k0 < K; k0 += BK) {
        // ---- stage A tile: 128x32, 16 elems/thread (4x vec4, coalesced) ----
        {
            int ar = tid >> 3;            // 0..31
            int kq = (tid & 7) * 4;       // 0..28
#pragma unroll
            for (int j = 0; j < 4; ++j) {
                int row = ar + 32 * j;
                const float* ap = A + (size_t)(m0 + row) * K + k0 + kq;
                if (k0 + BK < K) __builtin_prefetch(ap + BK, 0, 1); // global_prefetch_b8
#pragma unroll
                for (int i = 0; i < 4; ++i)
                    As[row][kq + i] = (__bf16)ap[i];
            }
        }
        // ---- stage B tile: 32(k) x 128(n), 16 consecutive n per thread ----
        {
            int kk = tid >> 3;            // 0..31
            int nq = (tid & 7) * 16;      // 0..112
            if (headMajor) {
#pragma unroll
                for (int j = 0; j < 16; ++j) {
                    int gn = n0 + nq + j;
                    __bf16 val = (__bf16)0.0f;
                    if (gn < N)
                        val = (__bf16)B[(size_t)(gn >> 6) * ((size_t)K * 64) +
                                        (size_t)(k0 + kk) * 64 + (gn & 63)];
                    Bs[nq + j][kk] = val;
                }
            } else {
                const float* bp = B + (size_t)(k0 + kk) * N + n0 + nq;
                if (k0 + BK < K) __builtin_prefetch(bp + (size_t)BK * N, 0, 1);
#pragma unroll
                for (int j = 0; j < 16; ++j) {
                    int gn = n0 + nq + j;
                    Bs[nq + j][kk] = (gn < N) ? (__bf16)bp[j] : (__bf16)0.0f;
                }
            }
        }
        __syncthreads();

        // ---- fragments per ISA 16-bit A 16x32 layout ----
        v16bf a[2];
#pragma unroll
        for (int mi = 0; mi < 2; ++mi) {
            const int arow = wm * 32 + mi * 16 + r;
#pragma unroll
            for (int i = 0; i < 16; ++i)
                a[mi][i] = As[arow][kb + i + (i >= 8 ? 8 : 0)];
        }
#pragma unroll
        for (int ni = 0; ni < 4; ++ni) {
            v16bf b;
            const int bn = wn * 64 + ni * 16 + r;
#pragma unroll
            for (int i = 0; i < 16; ++i)
                b[i] = Bs[bn][kb + i + (i >= 8 ? 8 : 0)];
#pragma unroll
            for (int mi = 0; mi < 2; ++mi)
                acc[mi][ni] = __builtin_amdgcn_wmma_f32_16x16x32_bf16(
                    false, a[mi], false, b, (short)0, acc[mi][ni], false, false);
        }
        __syncthreads();
    }

    // ---- epilogue: C layout  M = vgpr + 8*half, N = lane&15 ----
#pragma unroll
    for (int mi = 0; mi < 2; ++mi) {
        const int rowb = m0 + wm * 32 + mi * 16 + half * 8;
#pragma unroll
        for (int ni = 0; ni < 4; ++ni) {
            const int col = n0 + wn * 64 + ni * 16 + r;
            if (col < N) {
                const float bval = bias ? bias[col] : 0.0f;
#pragma unroll
                for (int rv = 0; rv < 8; ++rv) {
                    float o = acc[mi][ni][rv] + bval;
                    if (relu) o = fmaxf(o, 0.0f);
                    C[(size_t)(rowb + rv) * N + col] = o;
                }
            }
        }
    }
}

// ---------------------------------------------------------------------------
// Flash attention: one wave per (b, h, 16-query chunk); stream keys 32/step.
// q,k,v,o are [B,T,D] fp32 with head-major columns (col = h*64 + s).
// ---------------------------------------------------------------------------
__global__ __launch_bounds__(32) void attn_kernel(
    const float* __restrict__ q, const float* __restrict__ k,
    const float* __restrict__ v, float* __restrict__ o)
{
    const int blk = blockIdx.x;
    const int qc  = blk % (GT / 16);
    const int bh  = blk / (GT / 16);
    const int h   = bh % GH;
    const int b   = bh / GH;
    const int q0  = qc * 16;

    const int lane = threadIdx.x & 31;
    const int half = lane >> 4;
    const int r    = lane & 15;
    const int kb   = half * 8;

    const float* qb  = q + (size_t)b * GT * GD + h * GHS;
    const float* kbp = k + (size_t)b * GT * GD + h * GHS;
    const float* vb  = v + (size_t)b * GT * GD + h * GHS;
    float*       ob  = o + (size_t)b * GT * GD + h * GHS;

    // Q fragments (16 x 64 -> two 16x32 A-frags), loaded once
    v16bf qf[2];
#pragma unroll
    for (int c = 0; c < 2; ++c) {
        const int row = q0 + r;
#pragma unroll
        for (int i = 0; i < 16; ++i) {
            int s = c * 32 + kb + i + (i >= 8 ? 8 : 0);
            qf[c][i] = (__bf16)qb[(size_t)row * GD + s];
        }
    }

    float mrow[8], lrow[8];
    v8f accO[4];
#pragma unroll
    for (int rv = 0; rv < 8; ++rv) { mrow[rv] = -1e30f; lrow[rv] = 0.0f; }
#pragma unroll
    for (int t = 0; t < 4; ++t) { v8f z = {}; accO[t] = z; }

    __shared__ __bf16 Ps[16][32];

    const float scale = 0.125f;          // 1/sqrt(64)
    const int nch = qc / 2 + 1;          // 32-key chunks covering keys <= q0+15

    for (int kc = 0; kc < nch; ++kc) {
        const int c0 = kc * 32;

        // ---- S = Q @ K^T (two 16x16 tiles, K=64 -> 2 wmma each) ----
        v8f st[2];
        { v8f z = {}; st[0] = z; st[1] = z; }
#pragma unroll
        for (int u = 0; u < 2; ++u) {
            const int key = c0 + 16 * u + r;   // B-frag lane -> key column
#pragma unroll
            for (int c = 0; c < 2; ++c) {
                v16bf kf;
#pragma unroll
                for (int i = 0; i < 16; ++i) {
                    int s = c * 32 + kb + i + (i >= 8 ? 8 : 0);
                    kf[i] = (__bf16)kbp[(size_t)key * GD + s];
                }
                st[u] = __builtin_amdgcn_wmma_f32_16x16x32_bf16(
                            false, qf[c], false, kf, (short)0, st[u], false, false);
            }
        }

        // ---- scale + causal mask ----
#pragma unroll
        for (int u = 0; u < 2; ++u) {
            const int key = c0 + 16 * u + r;
#pragma unroll
            for (int rv = 0; rv < 8; ++rv) {
                int qrow = q0 + rv + 8 * half;
                float val = st[u][rv] * scale;
                if (key > qrow) val = -1e30f;
                st[u][rv] = val;
            }
        }

        // ---- online softmax: row max/sum across the 16-lane half ----
#pragma unroll
        for (int rv = 0; rv < 8; ++rv) {
            float m = fmaxf(st[0][rv], st[1][rv]);
#pragma unroll
            for (int d = 1; d < 16; d <<= 1)
                m = fmaxf(m, __shfl_xor(m, d, 32));
            float mn    = fmaxf(mrow[rv], m);
            float alpha = __expf(mrow[rv] - mn);
            mrow[rv] = mn;
            lrow[rv] *= alpha;
#pragma unroll
            for (int t = 0; t < 4; ++t) accO[t][rv] *= alpha;

            float p0 = __expf(st[0][rv] - mn);
            float p1 = __expf(st[1][rv] - mn);
            st[0][rv] = p0;
            st[1][rv] = p1;
            float sum = p0 + p1;
#pragma unroll
            for (int d = 1; d < 16; d <<= 1)
                sum += __shfl_xor(sum, d, 32);
            lrow[rv] += sum;
        }

        // ---- restage P (C-layout -> A-layout) through LDS ----
        __syncthreads();
#pragma unroll
        for (int u = 0; u < 2; ++u)
#pragma unroll
            for (int rv = 0; rv < 8; ++rv)
                Ps[rv + 8 * half][16 * u + r] = (__bf16)st[u][rv];
        __syncthreads();

        v16bf pf;
#pragma unroll
        for (int i = 0; i < 16; ++i)
            pf[i] = Ps[r][kb + i + (i >= 8 ? 8 : 0)];

        // ---- O += P @ V  (K=32 keys, four 16-wide s tiles) ----
#pragma unroll
        for (int t = 0; t < 4; ++t) {
            v16bf vf;
            const int s = 16 * t + r;
#pragma unroll
            for (int i = 0; i < 16; ++i) {
                int key = c0 + kb + i + (i >= 8 ? 8 : 0);
                vf[i] = (__bf16)vb[(size_t)key * GD + s];
            }
            accO[t] = __builtin_amdgcn_wmma_f32_16x16x32_bf16(
                          false, pf, false, vf, (short)0, accO[t], false, false);
        }
    }

    // ---- epilogue: O / l ----
#pragma unroll
    for (int t = 0; t < 4; ++t) {
        const int s = 16 * t + r;
#pragma unroll
        for (int rv = 0; rv < 8; ++rv) {
            int row = q0 + rv + 8 * half;
            ob[(size_t)row * GD + s] = accO[t][rv] / lrow[rv];
        }
    }
}

// ---------------------------------------------------------------------------
// Fused residual + LayerNorm:  xout = LN(xres + delta) * g + b   (row-wise)
// One block (256 thr) per row of 768; in-place safe (block owns its row).
// ---------------------------------------------------------------------------
__global__ __launch_bounds__(256) void ln_kernel(
    const float* __restrict__ xres, const float* __restrict__ delta,
    const float* __restrict__ g, const float* __restrict__ bta,
    float* __restrict__ xout)
{
    __shared__ float red[256];
    __shared__ float s_mean, s_rstd;
    const int row = blockIdx.x;
    const int tid = threadIdx.x;
    const float* xr = xres + (size_t)row * GD;
    const float* dl = delta + (size_t)row * GD;

    float vals[3];
    float sum = 0.0f;
#pragma unroll
    for (int j = 0; j < 3; ++j) {
        int i = tid + j * 256;
        vals[j] = xr[i] + dl[i];
        sum += vals[j];
    }
    red[tid] = sum;
    __syncthreads();
    for (int s = 128; s > 0; s >>= 1) {
        if (tid < s) red[tid] += red[tid + s];
        __syncthreads();
    }
    if (tid == 0) s_mean = red[0] * (1.0f / GD);
    __syncthreads();
    const float mean = s_mean;

    float vsum = 0.0f;
#pragma unroll
    for (int j = 0; j < 3; ++j) {
        float d2 = vals[j] - mean;
        vsum += d2 * d2;
    }
    red[tid] = vsum;
    __syncthreads();
    for (int s = 128; s > 0; s >>= 1) {
        if (tid < s) red[tid] += red[tid + s];
        __syncthreads();
    }
    if (tid == 0) s_rstd = rsqrtf(red[0] * (1.0f / GD) + 1e-5f);
    __syncthreads();
    const float rstd = s_rstd;

#pragma unroll
    for (int j = 0; j < 3; ++j) {
        int i = tid + j * 256;
        xout[(size_t)row * GD + i] = (vals[j] - mean) * rstd * g[i] + bta[i];
    }
}

// ---------------------------------------------------------------------------
// Orchestration
// ---------------------------------------------------------------------------
extern "C" void kernel_launch(void* const* d_in, const int* in_sizes, int n_in,
                              void* d_out, int out_size, void* d_ws, size_t ws_size,
                              hipStream_t stream)
{
    const int*   idx    = (const int*)  d_in[0];
    const float* tok    = (const float*)d_in[1];
    const float* pos    = (const float*)d_in[2];
    const float* wq     = (const float*)d_in[3];
    const float* wk     = (const float*)d_in[4];
    const float* wv     = (const float*)d_in[5];
    const float* wproj  = (const float*)d_in[6];
    const float* bproj  = (const float*)d_in[7];
    const float* w1     = (const float*)d_in[8];
    const float* b1     = (const float*)d_in[9];
    const float* w2     = (const float*)d_in[10];
    const float* b2     = (const float*)d_in[11];
    const float* ln1_g  = (const float*)d_in[12];
    const float* ln1_b  = (const float*)d_in[13];
    const float* ln2_g  = (const float*)d_in[14];
    const float* ln2_b  = (const float*)d_in[15];
    const float* lm_w   = (const float*)d_in[16];
    const float* lm_b   = (const float*)d_in[17];
    float* out = (float*)d_out;

    // workspace layout (floats): x, q, k, v, o, tmp  (each M*D) + h1 (M*4D)
    const size_t NMD = (size_t)GM * GD;
    float* x   = (float*)d_ws;
    float* q   = x   + NMD;
    float* k   = q   + NMD;
    float* v   = k   + NMD;
    float* o   = v   + NMD;
    float* tmp = o   + NMD;
    float* h1  = tmp + NMD;                  // M * 3072

    const size_t WPL_QKV  = (size_t)GH * GD * GHS;   // per-layer wq/wk/wv
    const size_t WPL_PROJ = (size_t)GD * GD;
    const size_t WPL_W1   = (size_t)GD * GFF;
    const size_t WPL_W2   = (size_t)GFF * GD;

    const dim3 gD  ((GD  + BN - 1) / BN, GM / BM);   // 6 x 16
    const dim3 gFF ((GFF + BN - 1) / BN, GM / BM);   // 24 x 16
    const dim3 gV  ((GV  + BN - 1) / BN, GM / BM);   // 393 x 16

    embed_kernel<<<(GM * GD + 255) / 256, 256, 0, stream>>>(idx, tok, pos, x);

    for (int l = 0; l < GL; ++l) {
        gemm_kernel<<<gD, 256, 0, stream>>>(x, wq + (size_t)l * WPL_QKV, nullptr,
                                            q, GM, GD, GD, 1, 0);
        gemm_kernel<<<gD, 256, 0, stream>>>(x, wk + (size_t)l * WPL_QKV, nullptr,
                                            k, GM, GD, GD, 1, 0);
        gemm_kernel<<<gD, 256, 0, stream>>>(x, wv + (size_t)l * WPL_QKV, nullptr,
                                            v, GM, GD, GD, 1, 0);

        attn_kernel<<<GB * GH * (GT / 16), 32, 0, stream>>>(q, k, v, o);

        gemm_kernel<<<gD, 256, 0, stream>>>(o, wproj + (size_t)l * WPL_PROJ,
                                            bproj + (size_t)l * GD,
                                            tmp, GM, GD, GD, 0, 0);
        ln_kernel<<<GM, 256, 0, stream>>>(x, tmp, ln1_g + (size_t)l * GD,
                                          ln1_b + (size_t)l * GD, x);

        gemm_kernel<<<gFF, 256, 0, stream>>>(x, w1 + (size_t)l * WPL_W1,
                                             b1 + (size_t)l * GFF,
                                             h1, GM, GFF, GD, 0, 1);
        gemm_kernel<<<gD, 256, 0, stream>>>(h1, w2 + (size_t)l * WPL_W2,
                                            b2 + (size_t)l * GD,
                                            tmp, GM, GD, GFF, 0, 0);
        ln_kernel<<<GM, 256, 0, stream>>>(x, tmp, ln2_g + (size_t)l * GD,
                                          ln2_b + (size_t)l * GD, x);
    }

    gemm_kernel<<<gV, 256, 0, stream>>>(x, lm_w, lm_b, out, GM, GV, GD, 0, 0);
}
